// BondMessagePassing_57921928954075
// MI455X (gfx1250) — matrix operand order repaired and checked by reading, not compiled, not run on hardware
//
#include <hip/hip_runtime.h>

// ---------------- problem constants ----------------
#define NN 200000      // nodes
#define NE 400000      // directed edges
#define DV 133
#define DE 14
#define DH 200
#define KI 160         // pad(DV+DE=147) to mult of 32
#define KH 224         // pad(DH=200)   to mult of 32
#define KO 352         // pad(DV+DH=333) to mult of 32
#define NP 208         // pad(DH) to mult of 16
#define NT 13          // NP / 16 output tiles per wave

// ---------------- types ----------------
typedef __attribute__((ext_vector_type(16))) __bf16 v16bf;
typedef __attribute__((ext_vector_type(8)))  float  v8f;
typedef __attribute__((ext_vector_type(4)))  int    v4i;

struct alignas(16) U4 { unsigned int x, y, z, w; };
struct U32x8 { U4 lo, hi; };

#if __has_builtin(__builtin_amdgcn_global_load_async_to_lds_b128) && \
    __has_builtin(__builtin_amdgcn_s_wait_asynccnt)
#define ASYNC_STAGE 1
#else
#define ASYNC_STAGE 0
#endif

static __device__ __forceinline__ unsigned short f2bf(float f) {
  unsigned int u = __builtin_bit_cast(unsigned int, f);
  u += 0x7fffu + ((u >> 16) & 1u);          // round-to-nearest-even
  return (unsigned short)(u >> 16);
}
static __device__ __forceinline__ float bf2f(unsigned short h) {
  return __builtin_bit_cast(float, (unsigned int)h << 16);
}
static __device__ __forceinline__ v16bf ld_frag2(const unsigned short* p0,
                                                 const unsigned short* p1) {
  U32x8 u;
  u.lo = *(const U4*)p0;
  u.hi = *(const U4*)p1;
  return __builtin_bit_cast(v16bf, u);
}

// Issue weight staging global->LDS. With gfx1250 async-LDS ops this runs on the
// ASYNC path (no VGPR round trip) and overlaps with the A-row gather; the wait
// is deferred to stage_wait() right before the block barrier.
static __device__ __forceinline__ void stage_issue(const unsigned short* __restrict__ Wt,
                                                   unsigned short* Bs, int n8,
                                                   int tid, int nth) {
#if ASYNC_STAGE
  for (int i = tid; i < n8; i += nth) {
    __builtin_amdgcn_global_load_async_to_lds_b128(
        (__attribute__((address_space(1))) v4i*)(Wt + i * 8),
        (__attribute__((address_space(3))) v4i*)(Bs + i * 8),
        0, 0);
  }
#else
  const U4* g = (const U4*)Wt;
  U4* s = (U4*)Bs;
  for (int i = tid; i < n8; i += nth) s[i] = g[i];
#endif
}
static __device__ __forceinline__ void stage_wait() {
#if ASYNC_STAGE
  __builtin_amdgcn_s_wait_asynccnt(0);
#endif
}

// One wave computes a 16 x 208 output tile (13 WMMA accumulators).
// As: [16][KPAD] bf16 activations (row-major).
// Bs: [NP][KPAD] bf16 weights, pre-transposed (row n = output col n over K).
// Fragment loads are double-buffered so ds_loads overlap the WMMAs.
template <int KPAD>
static __device__ __forceinline__ void wave_gemm(const unsigned short* As,
                                                 const unsigned short* Bs,
                                                 int lane, v8f acc[NT]) {
  const int half = lane >> 4;
  const int mrow = lane & 15;
  // 16-bit A 16x32 layout: lanes 0-15 hold K [kt+0..7] & [kt+16..23] of row mrow,
  // lanes 16-31 hold K [kt+8..15] & [kt+24..31].
  const unsigned short* arow = As + mrow * KPAD + half * 8;
  // B 32x16: same K layout as A (lanes 0-15 K [kt+0..7]&[kt+16..23], lanes 16-31 K [kt+8..15]&[kt+24..31]).
  const unsigned short* brow = Bs + mrow * KPAD + half * 8;
  v16bf a = ld_frag2(arow, arow + 16);
#pragma unroll
  for (int kt = 0; kt < KPAD; kt += 32) {
    v16bf b = ld_frag2(brow + kt, brow + kt + 16);
    v16bf an = a;
    if (kt + 32 < KPAD) {                 // prefetch next-K A fragment
      const unsigned short* pa = arow + kt + 32;
      an = ld_frag2(pa, pa + 16);
    }
#pragma unroll
    for (int nt = 0; nt < NT; ++nt) {
      v16bf bn = b;
      if (nt + 1 < NT) {                  // prefetch next-N B fragment
        const unsigned short* pb = brow + (nt + 1) * 16 * KPAD + kt;
        bn = ld_frag2(pb, pb + 16);
      }
      acc[nt] = __builtin_amdgcn_wmma_f32_16x16x32_bf16(
          false, a, false, b, (short)0, acc[nt], false, false);
      b = bn;
    }
    a = an;
  }
}

// ---------------- weight prep: f32 [K][DH] -> bf16 transposed [NP][KPAD] ----------------
__global__ void k_prep(const float* __restrict__ W, unsigned short* __restrict__ Wt,
                       int Kreal, int KPAD) {
  const int i = blockIdx.x * blockDim.x + threadIdx.x;
  if (i >= NP * KPAD) return;
  const int n = i / KPAD, k = i - n * KPAD;
  Wt[i] = f2bf((n < DH && k < Kreal) ? W[k * DH + n] : 0.f);
}

__global__ void k_zero(float* __restrict__ p, int n) {
  for (int i = blockIdx.x * blockDim.x + threadIdx.x; i < n;
       i += gridDim.x * blockDim.x)
    p[i] = 0.f;
}

// segment_sum: out[idx[e]] += H[e]  (f32 atomics into L2)
__global__ void k_scatter(const unsigned short* __restrict__ H,
                          const int* __restrict__ idx,
                          float* __restrict__ out) {
  const int e = blockIdx.x, k = threadIdx.x;
  if (k < DH) {
    const float v = bf2f(H[(long)e * DH + k]);
    __hip_atomic_fetch_add(&out[(long)idx[e] * DH + k], v,
                           __ATOMIC_RELAXED, __HIP_MEMORY_SCOPE_AGENT);
  }
}

// ---------------- GEMM 1: H0 = relu([V[src] || E] @ W_i + b_i) ----------------
__global__ void __launch_bounds__(256) k_gemm1(const float* __restrict__ V,
                                               const float* __restrict__ Ef,
                                               const int* __restrict__ src,
                                               const unsigned short* __restrict__ Wt,
                                               const float* __restrict__ bi,
                                               float* __restrict__ H0f,
                                               unsigned short* __restrict__ Hbf) {
  __shared__ unsigned short Bs[NP * KI];        // 66.5 KB
  __shared__ unsigned short As[8][16 * KI];     // 41 KB
  const int tid = threadIdx.x;
  stage_issue(Wt, Bs, NP * KI / 8, tid, 256);
  const int wave = tid >> 5, lane = tid & 31;
  const int ebase = (blockIdx.x * 8 + wave) * 16;
  unsigned short* A = As[wave];
  for (int r = 0; r < 16; ++r) {
    const int e = ebase + r;
    const float* vrow = V + (long)src[e] * DV;
    const float* erow = Ef + (long)e * DE;
    if (r < 15) __builtin_prefetch(V + (long)src[e + 1] * DV, 0, 3);
    auto elem = [&](int k) -> float {
      if (k < DV) return vrow[k];
      if (k < DV + DE) return erow[k - DV];
      return 0.f;
    };
    for (int k = lane * 2; k < KI; k += 64) {
      const unsigned int lo = f2bf(elem(k)), hi = f2bf(elem(k + 1));
      *(unsigned int*)(A + r * KI + k) = lo | (hi << 16);
    }
  }
  stage_wait();
  __syncthreads();
  v8f acc[NT];
  const v8f vz = {0.f, 0.f, 0.f, 0.f, 0.f, 0.f, 0.f, 0.f};
#pragma unroll
  for (int nt = 0; nt < NT; ++nt) acc[nt] = vz;
  wave_gemm<KI>(A, Bs, lane, acc);
  const int half = lane >> 4, mrow = lane & 15;
#pragma unroll
  for (int nt = 0; nt < NT; ++nt) {
    const int col = nt * 16 + mrow;
    if (col < DH) {
      const float b = bi[col];
#pragma unroll
      for (int r = 0; r < 8; ++r) {           // D layout: M = r + 8*half, N = mrow
        const int e = ebase + half * 8 + r;
        float v = acc[nt][r] + b;
        v = v > 0.f ? v : 0.f;
        H0f[(long)e * DH + col] = v;
        Hbf[(long)e * DH + col] = f2bf(v);
      }
    }
  }
}

// ------- GEMM 2: H = relu(H0 + (msg[src] - H[rev]) @ W_h + b_h), in place over Hbf -------
// rev[e] == e^1 here, so H[rev] lives inside this wave's own 16-edge tile -> safe in-place.
__global__ void __launch_bounds__(256) k_gemm2(const float* __restrict__ Msg,
                                               const int* __restrict__ src,
                                               const int* __restrict__ rev,
                                               const unsigned short* __restrict__ Wt,
                                               const float* __restrict__ bh,
                                               const float* __restrict__ H0f,
                                               unsigned short* __restrict__ Hbf) {
  __shared__ unsigned short Bs[NP * KH];        // 93 KB
  __shared__ unsigned short As[8][16 * KH];     // 57 KB
  const int tid = threadIdx.x;
  stage_issue(Wt, Bs, NP * KH / 8, tid, 256);
  const int wave = tid >> 5, lane = tid & 31;
  const int ebase = (blockIdx.x * 8 + wave) * 16;
  unsigned short* A = As[wave];
  for (int r = 0; r < 16; ++r) {
    const int e = ebase + r;
    const float* nm = Msg + (long)src[e] * DH;
    const int re = rev[e];
    const unsigned short* hr = Hbf + (long)re * DH;
    const bool sub = (re != e);
    if (r < 15) __builtin_prefetch(Msg + (long)src[e + 1] * DH, 0, 3);
    for (int k = lane * 2; k < KH; k += 64) {   // DH is even: pair never straddles
      float f0 = 0.f, f1 = 0.f;
      if (k < DH) {
        f0 = nm[k];
        f1 = nm[k + 1];
        if (sub) {
          f0 -= bf2f(hr[k]);
          f1 -= bf2f(hr[k + 1]);
        }
      }
      const unsigned int lo = f2bf(f0), hi = f2bf(f1);
      *(unsigned int*)(A + r * KH + k) = lo | (hi << 16);
    }
  }
  stage_wait();
  __syncthreads();
  v8f acc[NT];
  const v8f vz = {0.f, 0.f, 0.f, 0.f, 0.f, 0.f, 0.f, 0.f};
#pragma unroll
  for (int nt = 0; nt < NT; ++nt) acc[nt] = vz;
  wave_gemm<KH>(A, Bs, lane, acc);
  const int half = lane >> 4, mrow = lane & 15;
#pragma unroll
  for (int nt = 0; nt < NT; ++nt) {
    const int col = nt * 16 + mrow;
    if (col < DH) {
      const float b = bh[col];
#pragma unroll
      for (int r = 0; r < 8; ++r) {
        const int e = ebase + half * 8 + r;
        float v = acc[nt][r] + b + H0f[(long)e * DH + col];
        v = v > 0.f ? v : 0.f;
        Hbf[(long)e * DH + col] = f2bf(v);
      }
    }
  }
}

// ---------------- GEMM 3: out = relu([V || M_v] @ W_o + b_o) ----------------
__global__ void __launch_bounds__(128) k_gemm3(const float* __restrict__ V,
                                               const float* __restrict__ Mv,
                                               const unsigned short* __restrict__ Wt,
                                               const float* __restrict__ bo,
                                               float* __restrict__ out) {
  __shared__ unsigned short Bs[NP * KO];        // 146 KB
  __shared__ unsigned short As[4][16 * KO];     // 45 KB
  const int tid = threadIdx.x;
  stage_issue(Wt, Bs, NP * KO / 8, tid, 128);
  const int wave = tid >> 5, lane = tid & 31;
  const int nbase = (blockIdx.x * 4 + wave) * 16;
  unsigned short* A = As[wave];
  for (int r = 0; r < 16; ++r) {
    const int n = nbase + r;
    const float* vrow = V + (long)n * DV;
    const float* mrowp = Mv + (long)n * DH;
    if (r < 15) __builtin_prefetch(Mv + (long)(n + 1) * DH, 0, 3);
    auto elem = [&](int k) -> float {
      if (k < DV) return vrow[k];
      if (k < DV + DH) return mrowp[k - DV];
      return 0.f;
    };
    for (int k = lane * 2; k < KO; k += 64) {
      const unsigned int lo = f2bf(elem(k)), hi = f2bf(elem(k + 1));
      *(unsigned int*)(A + r * KO + k) = lo | (hi << 16);
    }
  }
  stage_wait();
  __syncthreads();
  v8f acc[NT];
  const v8f vz = {0.f, 0.f, 0.f, 0.f, 0.f, 0.f, 0.f, 0.f};
#pragma unroll
  for (int nt = 0; nt < NT; ++nt) acc[nt] = vz;
  wave_gemm<KO>(A, Bs, lane, acc);
  const int half = lane >> 4, mrow = lane & 15;
#pragma unroll
  for (int nt = 0; nt < NT; ++nt) {
    const int col = nt * 16 + mrow;
    if (col < DH) {
      const float b = bo[col];
#pragma unroll
      for (int r = 0; r < 8; ++r) {
        const int n = nbase + half * 8 + r;
        float v = acc[nt][r] + b;
        out[(long)n * DH + col] = v > 0.f ? v : 0.f;
      }
    }
  }
}

// ---------------- host orchestration ----------------
extern "C" void kernel_launch(void* const* d_in, const int* in_sizes, int n_in,
                              void* d_out, int out_size, void* d_ws, size_t ws_size,
                              hipStream_t stream) {
  const float* V  = (const float*)d_in[0];
  const float* Ef = (const float*)d_in[1];
  const int* ei   = (const int*)d_in[2];     // [2][NE]: row0=src, row1=dst
  const int* rev  = (const int*)d_in[3];
  const float* Wi = (const float*)d_in[4];
  const float* bi = (const float*)d_in[5];
  const float* Wh = (const float*)d_in[6];
  const float* bh = (const float*)d_in[7];
  const float* Wo = (const float*)d_in[8];
  const float* bo = (const float*)d_in[9];
  const int* src = ei;
  const int* dst = ei + NE;

  // workspace layout (all 256B-aligned)
  char* ws = (char*)d_ws;
  unsigned short* Hbf = (unsigned short*)ws;                       // NE*DH bf16   = 160 MB
  float* H0f = (float*)(ws + (size_t)NE * DH * 2);                 // NE*DH f32    = 320 MB
  float* Msg = (float*)(ws + (size_t)NE * DH * 2 + (size_t)NE * DH * 4); // NN*DH f32 = 160 MB
  unsigned short* WtI = (unsigned short*)((char*)Msg + (size_t)NN * DH * 4);
  unsigned short* WtH = WtI + NP * KI;
  unsigned short* WtO = WtH + NP * KH;

  k_prep<<<(NP * KI + 255) / 256, 256, 0, stream>>>(Wi, WtI, DV + DE, KI);
  k_prep<<<(NP * KH + 255) / 256, 256, 0, stream>>>(Wh, WtH, DH, KH);
  k_prep<<<(NP * KO + 255) / 256, 256, 0, stream>>>(Wo, WtO, DV + DH, KO);

  k_gemm1<<<NE / (16 * 8), 256, 0, stream>>>(V, Ef, src, WtI, bi, H0f, Hbf);

  for (int it = 0; it < 2; ++it) {
    k_zero<<<8192, 256, 0, stream>>>(Msg, NN * DH);
    k_scatter<<<NE, 256, 0, stream>>>(Hbf, dst, Msg);
    k_gemm2<<<NE / (16 * 8), 256, 0, stream>>>(Msg, src, rev, WtH, bh, H0f, Hbf);
  }

  k_zero<<<8192, 256, 0, stream>>>(Msg, NN * DH);
  k_scatter<<<NE, 256, 0, stream>>>(Hbf, src, Msg);
  k_gemm3<<<NN / (16 * 4), 128, 0, stream>>>(V, Msg, WtO, bo, (float*)d_out);
}
